// QuantizedLinear_57990648431159
// MI455X (gfx1250) — matrix operand order repaired and checked by reading, not compiled, not run on hardware
//
#include <hip/hip_runtime.h>
#include <stdint.h>

// ---- CDNA5 ext-vector types (trivially copyable, natural alignment) ----
typedef __attribute__((ext_vector_type(16))) _Float16 v16h;
typedef __attribute__((ext_vector_type(8)))  _Float16 v8h;
typedef __attribute__((ext_vector_type(8)))  float    v8f;
typedef __attribute__((ext_vector_type(4)))  float    f32x4;
typedef __attribute__((ext_vector_type(4)))  int      i32x4;

#define IN_F   4096
#define OUT_F  11008
#define TOKENS 4096

#define BM 128          // tokens per workgroup tile
#define BN 128          // out-features per workgroup tile
#define BK 32           // K per step == WMMA f16 K
#define NK (IN_F / BK)  // 128 K-steps
#define PADK 40         // padded LDS row: 80B (16B aligned, conflict-free frag reads)

// async global->LDS copy, 16 bytes per lane (GVS mode: sgpr64 base + vgpr32 offset)
#define ASYNC_B128(dst, voff, sbase)                                   \
    asm volatile("global_load_async_to_lds_b128 %0, %1, %2"            \
                 :: "v"(dst), "v"(voff), "s"(sbase) : "memory")
#define WAIT_ASYNC0() asm volatile("s_wait_asynccnt 0x0" ::: "memory")

// =====================================================================
// one-time conversion kernels (bandwidth-bound, single pass)
// =====================================================================
__global__ __launch_bounds__(256) void cvtA_kernel(const float* __restrict__ A,
                                                   _Float16* __restrict__ Af)
{
    const size_t i = ((size_t)blockIdx.x * 256 + threadIdx.x) * 8;
    f32x4 a0 = *reinterpret_cast<const f32x4*>(A + i);
    f32x4 a1 = *reinterpret_cast<const f32x4*>(A + i + 4);
    v8h h;
#pragma unroll
    for (int j = 0; j < 4; ++j) { h[j] = (_Float16)a0[j]; h[4 + j] = (_Float16)a1[j]; }
    *reinterpret_cast<v8h*>(Af + i) = h;
}

__global__ __launch_bounds__(256) void cvtW_kernel(const int* __restrict__ Wq,
                                                   _Float16* __restrict__ Wf)
{
    const size_t i = ((size_t)blockIdx.x * 256 + threadIdx.x) * 8;
    i32x4 w0 = *reinterpret_cast<const i32x4*>(Wq + i);
    i32x4 w1 = *reinterpret_cast<const i32x4*>(Wq + i + 4);
    v8h h;
#pragma unroll
    for (int j = 0; j < 4; ++j) { h[j] = (_Float16)w0[j]; h[4 + j] = (_Float16)w1[j]; }
    *reinterpret_cast<v8h*>(Wf + i) = h;
}

// deterministic per-token row sum of A (for algebraic zero-point correction)
__global__ __launch_bounds__(256) void rowsum_kernel(const float* __restrict__ A,
                                                     float* __restrict__ rs)
{
    __shared__ float red[256];
    const int row = blockIdx.x;
    const float* p = A + (size_t)row * IN_F;
    float s = 0.f;
    for (int j = threadIdx.x; j < IN_F; j += 256) s += p[j];
    red[threadIdx.x] = s;
    __syncthreads();
    for (int off = 128; off > 0; off >>= 1) {
        if (threadIdx.x < off) red[threadIdx.x] += red[threadIdx.x + off];
        __syncthreads();
    }
    if (threadIdx.x == 0) rs[row] = red[0];
}

// =====================================================================
// main path: f16 WMMA GEMM fed by async global->LDS copies
// out[t,o] = (sum_k A[t,k]*Wq[o,k] - rowsum[t]*zp[o]) * s[o] + bias[o]
// =====================================================================
__global__ __launch_bounds__(256)
void gemm_f16_async(const _Float16* __restrict__ Af,
                    const _Float16* __restrict__ Wf,
                    const float* __restrict__ rowsum,
                    const float* __restrict__ scales,
                    const float* __restrict__ zps,
                    const float* __restrict__ bias,
                    float* __restrict__ out)
{
    __shared__ __align__(16) _Float16 Ash[2][BM][PADK];
    __shared__ __align__(16) _Float16 Bsh[2][BN][PADK];

    const int tid  = threadIdx.x;
    const int lane = tid & 31;
    const int wave = tid >> 5;     // 0..7
    const int wm   = wave >> 2;    // 0..1  (M)
    const int wn   = wave & 3;     // 0..3  (N)
    const int half = lane >> 4;
    const int l16  = lane & 15;

    const int nTile = blockIdx.x * BN;
    const int mTile = blockIdx.y * BM;

    // staging: 2 threads per tile row, each owns a 32B chunk-pair of the 64B K-slice
    const int ldRow = tid >> 1;            // 0..127
    const int ldCh  = tid & 1;             // 0/1 -> byte 0 / 32 within row slice

    const uint64_t aBase  = (uint64_t)(uintptr_t)(Af + (size_t)mTile * IN_F);
    const uint64_t wBase  = (uint64_t)(uintptr_t)(Wf + (size_t)nTile * IN_F);
    const uint32_t srcOff = (uint32_t)ldRow * (IN_F * 2) + (uint32_t)ldCh * 32;

    // LDS byte addresses for this thread's two 16B chunks, per buffer
    const uint32_t aD0 = (uint32_t)(uintptr_t)&Ash[0][ldRow][ldCh * 16];
    const uint32_t aD1 = (uint32_t)(uintptr_t)&Ash[1][ldRow][ldCh * 16];
    const uint32_t wD0 = (uint32_t)(uintptr_t)&Bsh[0][ldRow][ldCh * 16];
    const uint32_t wD1 = (uint32_t)(uintptr_t)&Bsh[1][ldRow][ldCh * 16];

    // epilogue per-column constants
    float scl[2], bse[2], zpl[2];
#pragma unroll
    for (int nt = 0; nt < 2; ++nt) {
        const int col = nTile + wn * 32 + nt * 16 + l16;
        scl[nt] = scales[col];
        bse[nt] = bias[col];
        zpl[nt] = zps[col];
    }

    v8f acc[4][2] = {};

    // ---- prologue: async-fill buffer 0 ----
    ASYNC_B128(aD0,       srcOff,       aBase);
    ASYNC_B128(aD0 + 16u, srcOff + 16u, aBase);
    ASYNC_B128(wD0,       srcOff,       wBase);
    ASYNC_B128(wD0 + 16u, srcOff + 16u, wBase);
    WAIT_ASYNC0();
    __syncthreads();

    // ---- main K loop: async-fill next buffer while WMMAs chew current ----
    for (int ks = 0; ks < NK; ++ks) {
        const int buf = ks & 1;

        // warm L2 ahead of first touch (device-scope prefetch)
        if (ks + 6 < NK) {
            __builtin_prefetch((const char*)(uintptr_t)aBase + srcOff + (size_t)(ks + 6) * (BK * 2), 0, 2);
            __builtin_prefetch((const char*)(uintptr_t)wBase + srcOff + (size_t)(ks + 6) * (BK * 2), 0, 2);
        }

        if (ks + 1 < NK) {
            const uint64_t aB = aBase + (uint64_t)(ks + 1) * (BK * 2);
            const uint64_t wB = wBase + (uint64_t)(ks + 1) * (BK * 2);
            const uint32_t aD = buf ? aD0 : aD1;
            const uint32_t wD = buf ? wD0 : wD1;
            ASYNC_B128(aD,       srcOff,       aB);
            ASYNC_B128(aD + 16u, srcOff + 16u, aB);
            ASYNC_B128(wD,       srcOff,       wB);
            ASYNC_B128(wD + 16u, srcOff + 16u, wB);
        }

        // fragment loads from LDS per ISA §7.12.2 layouts
        v16h afr[4], bfr[2];
#pragma unroll
        for (int mt = 0; mt < 4; ++mt) {
            const int row = wm * 64 + mt * 16 + l16;     // lane holds row M=l16
            v8h lo = *reinterpret_cast<const v8h*>(&Ash[buf][row][half * 8]);
            v8h hi = *reinterpret_cast<const v8h*>(&Ash[buf][row][16 + half * 8]);
            afr[mt] = __builtin_shufflevector(lo, hi, 0,1,2,3,4,5,6,7,8,9,10,11,12,13,14,15);
        }
#pragma unroll
        for (int nt = 0; nt < 2; ++nt) {
            const int row = wn * 32 + nt * 16 + l16;     // lane holds column N=l16
            v8h lo = *reinterpret_cast<const v8h*>(&Bsh[buf][row][half * 8]);
            v8h hi = *reinterpret_cast<const v8h*>(&Bsh[buf][row][16 + half * 8]);
            bfr[nt] = __builtin_shufflevector(lo, hi, 0,1,2,3,4,5,6,7,8,9,10,11,12,13,14,15);
        }

#pragma unroll
        for (int mt = 0; mt < 4; ++mt)
#pragma unroll
            for (int nt = 0; nt < 2; ++nt)
                acc[mt][nt] = __builtin_amdgcn_wmma_f32_16x16x32_f16(
                    false, afr[mt], false, bfr[nt],
                    (short)0, acc[mt][nt], false, false);

        if (ks + 1 < NK) WAIT_ASYNC0();
        __syncthreads();
    }

    // ---- epilogue: zero-point correction, scale, bias ----
#pragma unroll
    for (int mt = 0; mt < 4; ++mt) {
        const size_t rbase = (size_t)(mTile + wm * 64 + mt * 16 + half * 8);
        float rs[8];
#pragma unroll
        for (int v = 0; v < 8; ++v) rs[v] = rowsum[rbase + v];
#pragma unroll
        for (int nt = 0; nt < 2; ++nt) {
            const int col = nTile + wn * 32 + nt * 16 + l16;
#pragma unroll
            for (int v = 0; v < 8; ++v)
                out[(rbase + v) * OUT_F + col] =
                    (acc[mt][nt][v] - rs[v] * zpl[nt]) * scl[nt] + bse[nt];
        }
    }
}

// =====================================================================
// fallback (no workspace): fused-conversion GEMM
// =====================================================================
__global__ __launch_bounds__(256)
void qlinear_fused(const float* __restrict__ A,
                   const int*   __restrict__ Wq,
                   const float* __restrict__ scales,
                   const float* __restrict__ zps,
                   const float* __restrict__ bias,
                   float* __restrict__ out)
{
    __shared__ __align__(16) _Float16 Ash[2][BM][PADK];
    __shared__ __align__(16) _Float16 Bsh[2][BN][PADK];

    const int tid  = threadIdx.x;
    const int lane = tid & 31;
    const int wave = tid >> 5;
    const int wm   = wave >> 2;
    const int wn   = wave & 3;
    const int half = lane >> 4;
    const int l16  = lane & 15;

    const int nTile = blockIdx.x * BN;
    const int mTile = blockIdx.y * BM;

    const int ldRow  = tid >> 1;
    const int ldHalf = tid & 1;
    const float* Aload = A  + (size_t)(mTile + ldRow) * IN_F + ldHalf * 16;
    const int*   Wload = Wq + (size_t)(nTile + ldRow) * IN_F + ldHalf * 16;
    const float  zp    = zps[nTile + ldRow];

    float scl[2], bse[2];
#pragma unroll
    for (int nt = 0; nt < 2; ++nt) {
        const int col = nTile + wn * 32 + nt * 16 + l16;
        scl[nt] = scales[col];
        bse[nt] = bias[col];
    }

    v8f acc[4][2] = {};

    {
        const f32x4* ap = reinterpret_cast<const f32x4*>(Aload);
        const i32x4* wp = reinterpret_cast<const i32x4*>(Wload);
        f32x4 a0 = ap[0], a1 = ap[1], a2 = ap[2], a3 = ap[3];
        i32x4 w0 = wp[0], w1 = wp[1], w2 = wp[2], w3 = wp[3];
        v8h alo, ahi, blo, bhi;
#pragma unroll
        for (int j = 0; j < 4; ++j) {
            alo[j]   = (_Float16)a0[j];  alo[4+j] = (_Float16)a1[j];
            ahi[j]   = (_Float16)a2[j];  ahi[4+j] = (_Float16)a3[j];
            blo[j]   = (_Float16)((float)w0[j] - zp);
            blo[4+j] = (_Float16)((float)w1[j] - zp);
            bhi[j]   = (_Float16)((float)w2[j] - zp);
            bhi[4+j] = (_Float16)((float)w3[j] - zp);
        }
        *reinterpret_cast<v8h*>(&Ash[0][ldRow][ldHalf * 16])     = alo;
        *reinterpret_cast<v8h*>(&Ash[0][ldRow][ldHalf * 16 + 8]) = ahi;
        *reinterpret_cast<v8h*>(&Bsh[0][ldRow][ldHalf * 16])     = blo;
        *reinterpret_cast<v8h*>(&Bsh[0][ldRow][ldHalf * 16 + 8]) = bhi;
    }
    __syncthreads();

    for (int ks = 0; ks < NK; ++ks) {
        const int buf = ks & 1;

        if (ks + 2 < NK) {
            __builtin_prefetch(Aload + (size_t)(ks + 2) * BK, 0, 2);
            __builtin_prefetch(Wload + (size_t)(ks + 2) * BK, 0, 2);
        }

        f32x4 a0, a1, a2, a3; i32x4 w0, w1, w2, w3;
        if (ks + 1 < NK) {
            const f32x4* ap = reinterpret_cast<const f32x4*>(Aload + (size_t)(ks + 1) * BK);
            const i32x4* wp = reinterpret_cast<const i32x4*>(Wload + (size_t)(ks + 1) * BK);
            a0 = ap[0]; a1 = ap[1]; a2 = ap[2]; a3 = ap[3];
            w0 = wp[0]; w1 = wp[1]; w2 = wp[2]; w3 = wp[3];
        }

        v16h afr[4], bfr[2];
#pragma unroll
        for (int mt = 0; mt < 4; ++mt) {
            const int row = wm * 64 + mt * 16 + l16;
            v8h lo = *reinterpret_cast<const v8h*>(&Ash[buf][row][half * 8]);
            v8h hi = *reinterpret_cast<const v8h*>(&Ash[buf][row][16 + half * 8]);
            afr[mt] = __builtin_shufflevector(lo, hi, 0,1,2,3,4,5,6,7,8,9,10,11,12,13,14,15);
        }
#pragma unroll
        for (int nt = 0; nt < 2; ++nt) {
            const int row = wn * 32 + nt * 16 + l16;
            v8h lo = *reinterpret_cast<const v8h*>(&Bsh[buf][row][half * 8]);
            v8h hi = *reinterpret_cast<const v8h*>(&Bsh[buf][row][16 + half * 8]);
            bfr[nt] = __builtin_shufflevector(lo, hi, 0,1,2,3,4,5,6,7,8,9,10,11,12,13,14,15);
        }

#pragma unroll
        for (int mt = 0; mt < 4; ++mt)
#pragma unroll
            for (int nt = 0; nt < 2; ++nt)
                acc[mt][nt] = __builtin_amdgcn_wmma_f32_16x16x32_f16(
                    false, afr[mt], false, bfr[nt],
                    (short)0, acc[mt][nt], false, false);

        if (ks + 1 < NK) {
            const int nbuf = buf ^ 1;
            v8h alo, ahi, blo, bhi;
#pragma unroll
            for (int j = 0; j < 4; ++j) {
                alo[j]   = (_Float16)a0[j];  alo[4+j] = (_Float16)a1[j];
                ahi[j]   = (_Float16)a2[j];  ahi[4+j] = (_Float16)a3[j];
                blo[j]   = (_Float16)((float)w0[j] - zp);
                blo[4+j] = (_Float16)((float)w1[j] - zp);
                bhi[j]   = (_Float16)((float)w2[j] - zp);
                bhi[4+j] = (_Float16)((float)w3[j] - zp);
            }
            *reinterpret_cast<v8h*>(&Ash[nbuf][ldRow][ldHalf * 16])     = alo;
            *reinterpret_cast<v8h*>(&Ash[nbuf][ldRow][ldHalf * 16 + 8]) = ahi;
            *reinterpret_cast<v8h*>(&Bsh[nbuf][ldRow][ldHalf * 16])     = blo;
            *reinterpret_cast<v8h*>(&Bsh[nbuf][ldRow][ldHalf * 16 + 8]) = bhi;
        }
        __syncthreads();
    }

#pragma unroll
    for (int mt = 0; mt < 4; ++mt) {
#pragma unroll
        for (int nt = 0; nt < 2; ++nt) {
            const int    col   = nTile + wn * 32 + nt * 16 + l16;
            const size_t rbase = (size_t)(mTile + wm * 64 + mt * 16 + half * 8);
#pragma unroll
            for (int v = 0; v < 8; ++v)
                out[(rbase + v) * OUT_F + col] = acc[mt][nt][v] * scl[nt] + bse[nt];
        }
    }
}

// =====================================================================
extern "C" void kernel_launch(void* const* d_in, const int* in_sizes, int n_in,
                              void* d_out, int out_size, void* d_ws, size_t ws_size,
                              hipStream_t stream) {
    (void)in_sizes; (void)n_in; (void)out_size;
    const float* A   = (const float*)d_in[0];   // [TOKENS, IN_F] fp32
    const int*   Wq  = (const int*)d_in[1];     // [OUT_F, IN_F] int8-in-int32
    const float* sc  = (const float*)d_in[2];   // [OUT_F]
    const float* zp  = (const float*)d_in[3];   // [OUT_F]
    const float* bi  = (const float*)d_in[4];   // [1, OUT_F]
    float*       out = (float*)d_out;           // [TOKENS, OUT_F] fp32

    const size_t A_BYTES  = (size_t)TOKENS * IN_F * sizeof(_Float16);
    const size_t W_BYTES  = (size_t)OUT_F * IN_F * sizeof(_Float16);
    const size_t RS_BYTES = (size_t)TOKENS * sizeof(float);

    dim3 grid(OUT_F / BN, TOKENS / BM);         // (86, 32)

    if (d_ws && ws_size >= A_BYTES + W_BYTES + RS_BYTES) {
        _Float16* Af = (_Float16*)d_ws;
        _Float16* Wf = (_Float16*)((char*)d_ws + A_BYTES);
        float*    rs = (float*)((char*)d_ws + A_BYTES + W_BYTES);

        cvtA_kernel<<<(unsigned)((size_t)TOKENS * IN_F / 8 / 256), 256, 0, stream>>>(A, Af);
        cvtW_kernel<<<(unsigned)((size_t)OUT_F * IN_F / 8 / 256), 256, 0, stream>>>(Wq, Wf);
        rowsum_kernel<<<TOKENS, 256, 0, stream>>>(A, rs);
        gemm_f16_async<<<grid, dim3(256), 0, stream>>>(Af, Wf, rs, sc, zp, bi, out);
    } else {
        qlinear_fused<<<grid, dim3(256), 0, stream>>>(A, Wq, sc, zp, bi, out);
    }
}
